// MMoEHANNet_49280454754814
// MI455X (gfx1250) — matrix operand order repaired and checked by reading, not compiled, not run on hardware
//
// MMoEHANNet for MI455X (gfx1250): bf16 WMMA GEMM pipeline + atomic segmented softmax.
// Dominant compute (~285 GFLOP dense GEMM) is done with v_wmma_f32_16x16x32_bf16,
// f32 accumulate; edge aggregation uses native f32 global atomics.
// Semantic attention (_semantic) over a single metapath is the identity -> kW/kb/qv unused.
#include <hip/hip_runtime.h>

typedef __bf16 bhalf;
typedef __attribute__((ext_vector_type(16))) __bf16 v16bf;
typedef __attribute__((ext_vector_type(8)))  __bf16 v8bf;
typedef __attribute__((ext_vector_type(8)))  float  v8f;

#define N_RNA   40000
#define N_DIS   3000
#define E_EDGES 600000
#define T_REL   10000
#define CDIV(a,b) (((a)+(b)-1)/(b))

// ---------------------------------------------------------------------------
// WMMA fragment loads (CDNA5 ISA 7.12.2 layouts, wave32).
// A (16x32 bf16): lane m (0-15) holds K k..k+7 and k+16..k+23; lanes 16-31 hold +8.
// B (32x16 bf16): lane n holds K k..k+7 and k+16..k+23; lanes 16-31 hold +8.
// Both operands are stored row-major [rows, K], so the GEMM computes A * B^T.
// ---------------------------------------------------------------------------
__device__ __forceinline__ v16bf frag_A(const bhalf* p) {
  v8bf lo = *(const v8bf*)p;
  v8bf hi = *(const v8bf*)(p + 16);
  v16bf f;
#pragma unroll
  for (int i = 0; i < 8; ++i) { f[i] = lo[i]; f[i + 8] = hi[i]; }
  return f;
}
__device__ __forceinline__ v16bf frag_B(const bhalf* p) {
  v8bf lo = *(const v8bf*)p;
  v8bf hi = *(const v8bf*)(p + 16);
  v16bf f;
#pragma unroll
  for (int i = 0; i < 8; ++i) { f[i] = lo[i]; f[i + 8] = hi[i]; }
  return f;
}

// mode 0: Cf = acc + bias                           (f32)
// mode 2: u[row, col&255] += rowscale[row*8+(col>>8)] * relu(acc + bias)   (MoE, atomic)
// mode 3: Cf = sigmoid(acc)                          (final scores)
// mode 4: Cb = bf16(acc + bias)
// mode 5: Cb = bf16(relu(acc + bias))
__global__ __launch_bounds__(256) void k_wmma_gemm(
    const bhalf* __restrict__ A, const bhalf* __restrict__ B,
    const float* __restrict__ bias, const float* __restrict__ rowscale,
    float* __restrict__ Cf, bhalf* __restrict__ Cb,
    int M, int K, int ldc, int Nstore, int mode)
{
  const int lane = threadIdx.x & 31;
  const int wave = threadIdx.x >> 5;          // 0..7
  const int wm = wave & 1, wn = wave >> 1;    // 2 x 4 wave grid
  const int m0 = blockIdx.x * 64 + wm * 32;
  const int n0 = blockIdx.y * 256 + wn * 64;

  v8f acc[2][4];
#pragma unroll
  for (int i = 0; i < 2; ++i)
#pragma unroll
    for (int j = 0; j < 4; ++j)
#pragma unroll
      for (int v = 0; v < 8; ++v) acc[i][j][v] = 0.f;

  const bhalf* Ab0 = A + (size_t)(m0 + (lane & 15)) * K + ((lane >> 4) << 3);
  const bhalf* Ab1 = Ab0 + (size_t)16 * K;
  const bhalf* Bb  = B + (size_t)(n0 + (lane & 15)) * K + ((lane >> 4) << 3);

  for (int k = 0; k < K; k += 32) {
    v16bf a0 = frag_A(Ab0 + k);
    v16bf a1 = frag_A(Ab1 + k);
#pragma unroll
    for (int j = 0; j < 4; ++j) {
      v16bf b = frag_B(Bb + (size_t)j * 16 * K + k);
      acc[0][j] = __builtin_amdgcn_wmma_f32_16x16x32_bf16(false, a0, false, b,
                                                          (short)0, acc[0][j], false, false);
      acc[1][j] = __builtin_amdgcn_wmma_f32_16x16x32_bf16(false, a1, false, b,
                                                          (short)0, acc[1][j], false, false);
    }
  }

  // C/D layout: lanes 0-15: VGPR v -> M=v; lanes 16-31: M=8+v; N = lane&15.
  const int nc = lane & 15;
  const int mo = (lane >> 4) << 3;
#pragma unroll
  for (int i = 0; i < 2; ++i)
#pragma unroll
    for (int j = 0; j < 4; ++j)
#pragma unroll
      for (int v = 0; v < 8; ++v) {
        int row = m0 + i * 16 + mo + v;
        int col = n0 + j * 16 + nc;
        if (row >= M || col >= Nstore) continue;
        float x = acc[i][j][v];
        if (bias) x += bias[col];
        size_t o = (size_t)row * ldc + col;
        if (mode == 0) {
          Cf[o] = x;
        } else if (mode == 2) {
          float rel = x > 0.f ? x : 0.f;
          atomicAdd(Cf + (size_t)row * 256 + (col & 255),
                    rowscale[(size_t)row * 8 + (col >> 8)] * rel);
        } else if (mode == 3) {
          Cf[o] = 1.f / (1.f + __expf(-x));
        } else if (mode == 4) {
          Cb[o] = (bhalf)x;
        } else { // 5
          x = x > 0.f ? x : 0.f;
          Cb[o] = (bhalf)x;
        }
      }
}

// ----------------------------- elementwise helpers -------------------------
__global__ void k_cvt(const float* __restrict__ src, bhalf* __restrict__ dst,
                      int srcRows, int dstRows, int cols, int relu)
{
  long long i = (long long)blockIdx.x * blockDim.x + threadIdx.x;
  long long total = (long long)dstRows * cols;
  if (i >= total) return;
  int r = (int)(i / cols);
  float v = (r < srcRows) ? src[i] : 0.f;
  if (relu) v = v > 0.f ? v : 0.f;
  dst[i] = (bhalf)v;
}

__global__ void k_cvtT(const float* __restrict__ src, bhalf* __restrict__ dst,
                       int Bc, int R, int C)
{ // src[b,r,c] -> dst[b,c,r]
  long long i = (long long)blockIdx.x * blockDim.x + threadIdx.x;
  long long total = (long long)Bc * R * C;
  if (i >= total) return;
  int b = (int)(i / ((long long)R * C));
  int j = (int)(i % ((long long)R * C));
  int c = j / R, r = j % R;
  dst[i] = (bhalf)src[((long long)b * R + r) * C + c];
}

__global__ void k_fill_f32(float* p, float v, long long n)
{
  long long i = (long long)blockIdx.x * blockDim.x + threadIdx.x;
  if (i < n) p[i] = v;
}
__global__ void k_fill_u32(unsigned int* p, unsigned int v, long long n)
{
  long long i = (long long)blockIdx.x * blockDim.x + threadIdx.x;
  if (i < n) p[i] = v;
}

// ----------------------------- HAN edge attention --------------------------
__global__ void k_headdot(const float* __restrict__ x, const float* __restrict__ a,
                          float* __restrict__ out, int n)
{ // out[i,h] = sum_d x[i, h*64+d] * a[h*64+d]
  int idx = blockIdx.x * blockDim.x + threadIdx.x;
  if (idx >= n * 8) return;
  int i = idx >> 3, h = idx & 7;
  const float* xp = x + (size_t)i * 512 + h * 64;
  const float* ap = a + h * 64;
  float s = 0.f;
  for (int d = 0; d < 64; ++d) s += xp[d] * ap[d];
  out[idx] = s;
}

__global__ void k_alpha_max(const float* __restrict__ as_, const float* __restrict__ ad_,
                            const int* __restrict__ esrc, const int* __restrict__ edst,
                            float* __restrict__ alpha, unsigned int* __restrict__ amax, int E)
{
  int idx = blockIdx.x * blockDim.x + threadIdx.x;
  if (idx >= E * 8) return;
  int e = idx >> 3, h = idx & 7;
  int s = esrc[e], d = edst[e];
  float v = as_[s * 8 + h] + ad_[d * 8 + h];
  v = v > 0.f ? v : 0.2f * v;                 // leaky_relu(0.2)
  alpha[idx] = v;
  unsigned int u = __float_as_uint(v);
  u = (u & 0x80000000u) ? ~u : (u | 0x80000000u); // order-preserving encode
  atomicMax(amax + (size_t)d * 8 + h, u);
}

__global__ void k_expden(float* __restrict__ alpha, const unsigned int* __restrict__ amax,
                         float* __restrict__ den, const int* __restrict__ edst, int E)
{
  int idx = blockIdx.x * blockDim.x + threadIdx.x;
  if (idx >= E * 8) return;
  int e = idx >> 3, h = idx & 7;
  int d = edst[e];
  unsigned int u = amax[(size_t)d * 8 + h];
  float mx = (u & 0x80000000u) ? __uint_as_float(u & 0x7fffffffu) : __uint_as_float(~u);
  float ex = __expf(alpha[idx] - mx);
  alpha[idx] = ex;
  atomicAdd(den + (size_t)d * 8 + h, ex);
}

__global__ void k_wnorm(float* __restrict__ alpha, const float* __restrict__ den,
                        const int* __restrict__ edst, int E)
{
  int idx = blockIdx.x * blockDim.x + threadIdx.x;
  if (idx >= E * 8) return;
  int e = idx >> 3, h = idx & 7;
  alpha[idx] = alpha[idx] / (den[(size_t)edst[e] * 8 + h] + 1e-16f);
}

__global__ void k_scatter(const float* __restrict__ xsrc, const float* __restrict__ w,
                          const int* __restrict__ esrc, const int* __restrict__ edst,
                          float* __restrict__ out, long long total)
{
  long long i = (long long)blockIdx.x * blockDim.x + threadIdx.x;
  if (i >= total) return;
  int e = (int)(i >> 9);
  int j = (int)(i & 511);
  int h = j >> 6;
  float val = xsrc[(size_t)esrc[e] * 512 + j] * w[(size_t)e * 8 + h];
  atomicAdd(out + (size_t)edst[e] * 512 + j, val);
}

// ----------------------------- MoE / heads glue ----------------------------
__global__ void k_colmean(const float* __restrict__ h, float* __restrict__ out,
                          int rows, int cols)
{
  int c = blockIdx.x * blockDim.x + threadIdx.x;
  if (c >= cols) return;
  float s = 0.f;
  for (int r = 0; r < rows; ++r) s += h[(size_t)r * cols + c];
  out[c] = s / (float)rows;
}

__global__ void k_scores(const float* __restrict__ hrna, const int* __restrict__ idx_rels,
                         const float* __restrict__ Wg, float* __restrict__ sc)
{ // sc[row,e] = sum_h h_rna[idx[row],h] * W_gate[row/T, h, e]
  int t = blockIdx.x * blockDim.x + threadIdx.x;
  if (t >= 4 * T_REL * 8) return;
  int row = t >> 3, e = t & 7;
  int r = row / T_REL;
  int n = idx_rels[row];
  const float* hp = hrna + (size_t)n * 512;
  const float* wp = Wg + (size_t)r * 512 * 8 + e;
  float s = 0.f;
  for (int h = 0; h < 512; ++h) s += hp[h] * wp[h * 8];
  sc[t] = s;
}

__global__ void k_build_fusion(const float* __restrict__ hrna, const float* __restrict__ dglob,
                               const int* __restrict__ idx_rels, bhalf* __restrict__ Ag)
{ // Ag[row, f] = f<512 ? h_rna[idx[row], f] : d_global[f-512]
  long long i = (long long)blockIdx.x * blockDim.x + threadIdx.x;
  if (i >= (long long)4 * T_REL * 1024) return;
  int row = (int)(i >> 10), f = (int)(i & 1023);
  float v = (f < 512) ? hrna[(size_t)idx_rels[row] * 512 + f] : dglob[f - 512];
  Ag[i] = (bhalf)v;
}

__global__ void k_umean(const float* __restrict__ u, float* __restrict__ U)
{ // U[r,d] = mean_t u[(r*T+t)*256 + d]
  int r = blockIdx.x, d = threadIdx.x;
  float s = 0.f;
  for (int t = 0; t < T_REL; ++t) s += u[((size_t)r * T_REL + t) * 256 + d];
  U[r * 256 + d] = s / (float)T_REL;
}

__global__ void k_crossattn(const float* __restrict__ U, const float* __restrict__ in_w,
                            const float* __restrict__ in_b, const float* __restrict__ out_w,
                            const float* __restrict__ out_b, float* __restrict__ Up)
{ // tiny 4-token, 8-head MHA; single 256-thread block
  __shared__ float sU[1024], sq[1024], sk[1024], sv[1024], so[1024], sattn[128];
  int t = threadIdx.x;
  for (int i = t; i < 1024; i += 256) sU[i] = U[i];
  __syncthreads();
  for (int i = t; i < 4 * 768; i += 256) {
    int tok = i / 768, o = i % 768;
    float s = in_b[o];
    const float* w = in_w + (size_t)o * 256;
    for (int c = 0; c < 256; ++c) s += sU[tok * 256 + c] * w[c];
    if (o < 256) sq[tok * 256 + o] = s;
    else if (o < 512) sk[tok * 256 + o - 256] = s;
    else sv[tok * 256 + o - 512] = s;
  }
  __syncthreads();
  if (t < 32) {
    int h = t >> 2, tq = t & 3;
    float sc[4]; float mx = -1e30f;
    for (int s = 0; s < 4; ++s) {
      float d = 0.f;
      for (int c = 0; c < 32; ++c) d += sq[tq * 256 + h * 32 + c] * sk[s * 256 + h * 32 + c];
      sc[s] = d * 0.17677669529663687f;   // 1/sqrt(32)
      if (sc[s] > mx) mx = sc[s];
    }
    float sum = 0.f;
    for (int s = 0; s < 4; ++s) { sc[s] = __expf(sc[s] - mx); sum += sc[s]; }
    for (int s = 0; s < 4; ++s) sattn[(h * 4 + tq) * 4 + s] = sc[s] / sum;
  }
  __syncthreads();
  for (int i = t; i < 1024; i += 256) {
    int tok = i >> 8, d = i & 255, h = d >> 5;
    float s = 0.f;
    for (int si = 0; si < 4; ++si) s += sattn[(h * 4 + tok) * 4 + si] * sv[si * 256 + d];
    so[i] = s;
  }
  __syncthreads();
  for (int i = t; i < 1024; i += 256) {
    int tok = i >> 8, d = i & 255;
    float s = out_b[d];
    const float* w = out_w + (size_t)d * 256;
    for (int c = 0; c < 256; ++c) s += so[tok * 256 + c] * w[c];
    Up[i] = s;
  }
}

__global__ void k_build_cat(const float* __restrict__ u, const float* __restrict__ Up,
                            bhalf* __restrict__ cat)
{ // cat[row, c] = c<256 ? u[row,c] : U_prime[row/T, c-256]
  long long i = (long long)blockIdx.x * blockDim.x + threadIdx.x;
  if (i >= (long long)4 * T_REL * 512) return;
  int row = (int)(i >> 9), c = (int)(i & 511);
  float v = (c < 256) ? u[(size_t)row * 256 + c] : Up[(row / T_REL) * 256 + (c - 256)];
  cat[i] = (bhalf)v;
}

// ---------------------------------------------------------------------------
extern "C" void kernel_launch(void* const* d_in, const int* in_sizes, int n_in,
                              void* d_out, int out_size, void* d_ws, size_t ws_size,
                              hipStream_t stream)
{
  (void)in_sizes; (void)n_in; (void)out_size; (void)ws_size;
  const float* x_rna    = (const float*)d_in[0];
  const float* x_dis    = (const float*)d_in[1];
  const float* Wd       = (const float*)d_in[2];
  const float* bd       = (const float*)d_in[3];
  const float* Wp_r     = (const float*)d_in[4];
  const float* bp_r     = (const float*)d_in[5];
  const float* Wp_d     = (const float*)d_in[6];
  const float* bp_d     = (const float*)d_in[7];
  const float* a_src_rd = (const float*)d_in[8];
  const float* a_dst_rd = (const float*)d_in[9];
  const float* a_src_dr = (const float*)d_in[10];
  const float* a_dst_dr = (const float*)d_in[11];
  // d_in[12..14] = kW, kb, qv: dead (semantic attention over 1 metapath == identity)
  const float* Wl       = (const float*)d_in[15];
  const float* bl       = (const float*)d_in[16];
  const float* W_exp    = (const float*)d_in[17];
  const float* b_exp    = (const float*)d_in[18];
  const float* W_gate   = (const float*)d_in[19];
  const float* W_head   = (const float*)d_in[20];
  const float* b_head   = (const float*)d_in[21];
  const float* in_w     = (const float*)d_in[22];
  const float* in_b     = (const float*)d_in[23];
  const float* out_w    = (const float*)d_in[24];
  const float* out_b    = (const float*)d_in[25];
  const float* Wdp      = (const float*)d_in[26];
  const float* bdp      = (const float*)d_in[27];
  const int* ei_rd_src  = (const int*)d_in[28];
  const int* ei_rd_dst  = (const int*)d_in[29];
  const int* ei_dr_src  = (const int*)d_in[30];
  const int* ei_dr_dst  = (const int*)d_in[31];
  const int* idx_rels   = (const int*)d_in[32];
  float* outp = (float*)d_out;

  // ---- workspace bump allocator (rows padded for tile overrun safety) ----
  char* wsb = (char*)d_ws;
  size_t off = 0;
  auto alloc = [&](size_t bytes) -> void* {
    void* p = wsb + off;
    off += (bytes + 255) & ~(size_t)255;
    return p;
  };
  bhalf* xdis_bf   = (bhalf*)alloc((size_t)3072 * 256 * 2);
  bhalf* Wd_bf     = (bhalf*)alloc((size_t)512 * 256 * 2);
  bhalf* xd_bf     = (bhalf*)alloc((size_t)3072 * 512 * 2);
  bhalf* xrna_bf   = (bhalf*)alloc((size_t)40064 * 512 * 2);
  bhalf* Wpr_bf    = (bhalf*)alloc((size_t)512 * 512 * 2);
  bhalf* Wpd_bf    = (bhalf*)alloc((size_t)512 * 512 * 2);
  bhalf* Wl_bf     = (bhalf*)alloc((size_t)512 * 512 * 2);
  bhalf* Wdp_bf    = (bhalf*)alloc((size_t)256 * 512 * 2);
  bhalf* WexpT_bf  = (bhalf*)alloc((size_t)2048 * 1024 * 2);
  bhalf* WheadT_bf = (bhalf*)alloc((size_t)1024 * 512 * 2);
  float* hr0       = (float*)alloc((size_t)40064 * 512 * 4);  // later reused as h_rna
  float* hd0       = (float*)alloc((size_t)3072 * 512 * 4);   // later reused as h_dis
  float* as_rd     = (float*)alloc((size_t)N_RNA * 8 * 4);
  float* ad_rd     = (float*)alloc((size_t)N_DIS * 8 * 4);
  float* as_dr     = (float*)alloc((size_t)N_DIS * 8 * 4);
  float* ad_dr     = (float*)alloc((size_t)N_RNA * 8 * 4);
  float* alpha     = (float*)alloc((size_t)E_EDGES * 8 * 4);  // shared rd/dr (sequential)
  unsigned int* amax = (unsigned int*)alloc((size_t)N_RNA * 8 * 4);
  float* den       = (float*)alloc((size_t)N_RNA * 8 * 4);
  float* outdis_f  = (float*)alloc((size_t)3072 * 512 * 4);
  float* outrna_f  = (float*)alloc((size_t)40064 * 512 * 4);  // later reused as u [40000,256]
  bhalf* outdis_bf = (bhalf*)alloc((size_t)3072 * 512 * 2);
  bhalf* outrna_bf = (bhalf*)alloc((size_t)40064 * 512 * 2);
  float* dglob     = (float*)alloc((size_t)512 * 4);
  float* scoresb   = (float*)alloc((size_t)4 * T_REL * 8 * 4);
  bhalf* Ag_bf     = (bhalf*)alloc((size_t)40064 * 1024 * 2);
  float* Umat      = (float*)alloc((size_t)4 * 256 * 4);
  float* Uprime    = (float*)alloc((size_t)4 * 256 * 4);
  bhalf* cat_bf    = (bhalf*)alloc((size_t)40064 * 512 * 2);
  bhalf* p_bf      = (bhalf*)alloc((size_t)40064 * 256 * 2);
  bhalf* hdis_bf   = (bhalf*)alloc((size_t)3072 * 512 * 2);
  bhalf* dproj_bf  = (bhalf*)alloc((size_t)3072 * 256 * 2);
  float* u = outrna_f; // alias: outrna_f is dead after cvt->outrna_bf

  const int TB = 256;
  auto g1 = [&](long long n) { return dim3((unsigned)CDIV(n, TB)); };
  auto gemm = [&](const bhalf* A, const bhalf* B, const float* bias, const float* rs,
                  float* Cf, bhalf* Cb, int M, int Npad, int K, int ldc, int Nstore, int mode) {
    dim3 g(CDIV(M, 64), Npad / 256);
    k_wmma_gemm<<<g, 256, 0, stream>>>(A, B, bias, rs, Cf, Cb, M, K, ldc, Nstore, mode);
  };

  // ---- bf16 conversions of inputs/weights ----
  k_cvt<<<g1(3008LL * 256), TB, 0, stream>>>(x_dis, xdis_bf, N_DIS, 3008, 256, 0);
  k_cvt<<<g1(512LL * 256), TB, 0, stream>>>(Wd, Wd_bf, 512, 512, 256, 0);
  k_cvt<<<g1(40000LL * 512), TB, 0, stream>>>(x_rna, xrna_bf, N_RNA, N_RNA, 512, 0);
  k_cvt<<<g1(512LL * 512), TB, 0, stream>>>(Wp_r, Wpr_bf, 512, 512, 512, 0);
  k_cvt<<<g1(512LL * 512), TB, 0, stream>>>(Wp_d, Wpd_bf, 512, 512, 512, 0);
  k_cvt<<<g1(512LL * 512), TB, 0, stream>>>(Wl, Wl_bf, 512, 512, 512, 0);
  k_cvt<<<g1(256LL * 512), TB, 0, stream>>>(Wdp, Wdp_bf, 256, 256, 512, 0);
  k_cvtT<<<g1(8LL * 1024 * 256), TB, 0, stream>>>(W_exp, WexpT_bf, 8, 1024, 256);
  k_cvtT<<<g1(4LL * 512 * 256), TB, 0, stream>>>(W_head, WheadT_bf, 4, 512, 256);

  // ---- HAN projections ----
  // xd = x_dis @ Wd^T + bd   -> bf16 directly (feeds next GEMM)
  gemm(xdis_bf, Wd_bf, bd, nullptr, nullptr, xd_bf, 3008, 512, 256, 512, 512, 4);
  // hr0 = x_rna @ Wp_r^T + bp_r
  gemm(xrna_bf, Wpr_bf, bp_r, nullptr, hr0, nullptr, N_RNA, 512, 512, 512, 512, 0);
  // hd0 = xd @ Wp_d^T + bp_d
  gemm(xd_bf, Wpd_bf, bp_d, nullptr, hd0, nullptr, 3008, 512, 512, 512, 512, 0);

  // ---- edge attention, direction rna->dis ----
  k_headdot<<<g1((long long)N_RNA * 8), TB, 0, stream>>>(hr0, a_src_rd, as_rd, N_RNA);
  k_headdot<<<g1((long long)N_DIS * 8), TB, 0, stream>>>(hd0, a_dst_rd, ad_rd, N_DIS);
  k_fill_u32<<<g1((long long)N_DIS * 8), TB, 0, stream>>>(amax, 0x00800000u, (long long)N_DIS * 8);
  k_fill_f32<<<g1((long long)N_DIS * 8), TB, 0, stream>>>(den, 0.f, (long long)N_DIS * 8);
  k_fill_f32<<<g1((long long)N_DIS * 512), TB, 0, stream>>>(outdis_f, 0.f, (long long)N_DIS * 512);
  k_alpha_max<<<g1((long long)E_EDGES * 8), TB, 0, stream>>>(as_rd, ad_rd, ei_rd_src, ei_rd_dst, alpha, amax, E_EDGES);
  k_expden<<<g1((long long)E_EDGES * 8), TB, 0, stream>>>(alpha, amax, den, ei_rd_dst, E_EDGES);
  k_wnorm<<<g1((long long)E_EDGES * 8), TB, 0, stream>>>(alpha, den, ei_rd_dst, E_EDGES);
  k_scatter<<<g1((long long)E_EDGES * 512), TB, 0, stream>>>(hr0, alpha, ei_rd_src, ei_rd_dst, outdis_f, (long long)E_EDGES * 512);

  // ---- edge attention, direction dis->rna ----
  k_headdot<<<g1((long long)N_DIS * 8), TB, 0, stream>>>(hd0, a_src_dr, as_dr, N_DIS);
  k_headdot<<<g1((long long)N_RNA * 8), TB, 0, stream>>>(hr0, a_dst_dr, ad_dr, N_RNA);
  k_fill_u32<<<g1((long long)N_RNA * 8), TB, 0, stream>>>(amax, 0x00800000u, (long long)N_RNA * 8);
  k_fill_f32<<<g1((long long)N_RNA * 8), TB, 0, stream>>>(den, 0.f, (long long)N_RNA * 8);
  k_fill_f32<<<g1((long long)N_RNA * 512), TB, 0, stream>>>(outrna_f, 0.f, (long long)N_RNA * 512);
  k_alpha_max<<<g1((long long)E_EDGES * 8), TB, 0, stream>>>(as_dr, ad_dr, ei_dr_src, ei_dr_dst, alpha, amax, E_EDGES);
  k_expden<<<g1((long long)E_EDGES * 8), TB, 0, stream>>>(alpha, amax, den, ei_dr_dst, E_EDGES);
  k_wnorm<<<g1((long long)E_EDGES * 8), TB, 0, stream>>>(alpha, den, ei_dr_dst, E_EDGES);
  k_scatter<<<g1((long long)E_EDGES * 512), TB, 0, stream>>>(hd0, alpha, ei_dr_src, ei_dr_dst, outrna_f, (long long)E_EDGES * 512);

  // ---- relu + final shared linear (semantic attention is identity) ----
  k_cvt<<<g1(3008LL * 512), TB, 0, stream>>>(outdis_f, outdis_bf, N_DIS, 3008, 512, 1);
  k_cvt<<<g1(40000LL * 512), TB, 0, stream>>>(outrna_f, outrna_bf, N_RNA, N_RNA, 512, 1);
  gemm(outrna_bf, Wl_bf, bl, nullptr, hr0, nullptr, N_RNA, 512, 512, 512, 512, 0); // h_rna
  gemm(outdis_bf, Wl_bf, bl, nullptr, hd0, nullptr, 3008, 512, 512, 512, 512, 0);  // h_dis
  float* h_rna = hr0;
  float* h_dis = hd0;

  // ---- MoE with per-relation gating ----
  k_colmean<<<g1(512), TB, 0, stream>>>(h_dis, dglob, N_DIS, 512);
  k_scores<<<g1(4LL * T_REL * 8), TB, 0, stream>>>(h_rna, idx_rels, W_gate, scoresb);
  k_build_fusion<<<g1(4LL * T_REL * 1024), TB, 0, stream>>>(h_rna, dglob, idx_rels, Ag_bf);
  k_fill_f32<<<g1(40000LL * 256), TB, 0, stream>>>(u, 0.f, 40000LL * 256);
  // all 8 experts fused as one N=2048 GEMM; epilogue: u += score * relu(.)
  gemm(Ag_bf, WexpT_bf, b_exp, scoresb, u, nullptr, 40000, 2048, 1024, 256, 2048, 2);

  // ---- cross attention over 4 task tokens ----
  k_umean<<<dim3(4), 256, 0, stream>>>(u, Umat);
  k_crossattn<<<dim3(1), 256, 0, stream>>>(Umat, in_w, in_b, out_w, out_b, Uprime);
  k_build_cat<<<g1(4LL * T_REL * 512), TB, 0, stream>>>(u, Uprime, cat_bf);

  // ---- per-relation heads: p = relu(cat @ W_head[r] + b_head[r]) -> bf16 ----
  for (int r = 0; r < 4; ++r) {
    gemm(cat_bf + (size_t)r * T_REL * 512, WheadT_bf + (size_t)r * 256 * 512,
         b_head + (size_t)r * 256, nullptr, nullptr, p_bf + (size_t)r * T_REL * 256,
         T_REL, 256, 512, 256, 256, 5);
  }

  // ---- disease projection + final sigmoid scores ----
  k_cvt<<<g1(3008LL * 512), TB, 0, stream>>>(h_dis, hdis_bf, N_DIS, 3008, 512, 0);
  gemm(hdis_bf, Wdp_bf, bdp, nullptr, nullptr, dproj_bf, 3008, 256, 512, 256, 256, 4);
  // out[4*T, 3000] = sigmoid(p @ d_proj^T); 480MB stored once, sigmoid fused in epilogue
  gemm(p_bf, dproj_bf, nullptr, nullptr, outp, nullptr, 40000, 3072, 256, 3000, 3000, 3);
}
